// GraphNet_3521873183411
// MI455X (gfx1250) — matrix operand order repaired and checked by reading, not compiled, not run on hardware
//
#include <hip/hip_runtime.h>
#include <hip/hip_fp16.h>
#include <hip/hip_bf16.h>

#define B_   16
#define N_   256
#define IN_  8
#define H_   32
#define NH_  4
#define DH_  8
#define SX   33   // padded LDS stride for 32-wide matrices
#define SH   65   // padded LDS stride for 64-wide matrices

typedef __attribute__((ext_vector_type(16))) _Float16 v16h;
typedef __attribute__((ext_vector_type(8)))  float    v8f;

struct LayerParams {
  const float *Wq,*bq,*Wk,*bk,*Wv,*bv,*Wo,*bo,*Wf1,*bf1,*Wf2,*bf2,*g1,*lb1,*g2,*lb2;
};

// ---------------------------------------------------------------------------
// WMMA GEMM on LDS activations: OUT[256,Nc] = X[256,Kd] @ W[Kd,Nc] + bias
// X, OUT in LDS (f32, padded stride); W, bias in global (f32, tiny, L2-hot).
// A/B converted to f16 at fragment build; accumulate f32 via
// v_wmma_f32_16x16x32_f16. 8 waves cover 16*(Nc/16) tiles.
// ---------------------------------------------------------------------------
__device__ void gemm_wmma(const float* __restrict__ X, int sx,
                          const float* __restrict__ W, const float* __restrict__ bias,
                          float* __restrict__ OUT, int so,
                          int Kd, int Nc, bool relu)
{
  const int lane  = threadIdx.x & 31;
  const int wave  = threadIdx.x >> 5;
  const int nT    = Nc >> 4;
  const int total = 16 * nT;              // 16 M-tiles of the 256 rows
  const int mloc  = lane & 15;
  const int hiK   = (lane >> 4) << 3;     // A: +8 K-offset & C: +8 row for upper half-wave
  const int koff  = (lane >> 4) << 4;     // unused; B loads use hiK (+8 K-offset), same k order as A

  for (int tile = wave; tile < total; tile += 8) {
    const int mt = tile / nT, nt = tile % nT;
    const int m  = mt * 16 + mloc;
    const int n  = nt * 16 + mloc;
    v8f c = {0.f,0.f,0.f,0.f,0.f,0.f,0.f,0.f};
    for (int k0 = 0; k0 < Kd; k0 += 32) {
      v16h a, b;
#pragma unroll
      for (int i = 0; i < 8; ++i) {
        a[i]     = (_Float16)X[m * sx + k0 + hiK + i];
        a[i + 8] = (_Float16)X[m * sx + k0 + 16 + hiK + i];
      }
#pragma unroll
      for (int i = 0; i < 8; ++i) {
        b[i]     = (_Float16)W[(k0 + hiK + i) * Nc + n];
        b[i + 8] = (_Float16)W[(k0 + 16 + hiK + i) * Nc + n];
      }
      c = __builtin_amdgcn_wmma_f32_16x16x32_f16(false, a, false, b,
                                                 (short)0, c, false, false);
    }
    const float bv = bias[n];
#pragma unroll
    for (int g = 0; g < 8; ++g) {
      const int row = mt * 16 + g + hiK;
      float v = c[g] + bv;
      if (relu) v = fmaxf(v, 0.f);
      OUT[row * so + n] = v;
    }
  }
}

// ---------------------------------------------------------------------------
// One transformer encoder layer fully in LDS. Xs updated in place.
// Buffers: Qs,Ks,Vs,As are 256*SX scratch; FF hidden [256,SH] aliases Qs..Ks.
// ---------------------------------------------------------------------------
__device__ void encoder_layer(float* Xs, float* Qs, float* Ks, float* Vs, float* As,
                              const float* mask, const LayerParams p)
{
  const int t = threadIdx.x;

  gemm_wmma(Xs, SX, p.Wq, p.bq, Qs, SX, 32, 32, false);
  gemm_wmma(Xs, SX, p.Wk, p.bk, Ks, SX, 32, 32, false);
  gemm_wmma(Xs, SX, p.Wv, p.bv, Vs, SX, 32, 32, false);
  __syncthreads();

  // Streaming 2-pass softmax attention; thread t = query row, DH=8 per head.
  const float scale = 0.3535533905932738f;   // 1/sqrt(8)
  for (int h = 0; h < NH_; ++h) {
    float q[DH_];
#pragma unroll
    for (int d = 0; d < DH_; ++d) q[d] = Qs[t * SX + h * DH_ + d];
    float mx = -1e30f;
    for (int k = 0; k < N_; ++k) {
      float sc;
      if (mask[k] < 0.5f) sc = -1e9f;
      else {
        sc = 0.f;
#pragma unroll
        for (int d = 0; d < DH_; ++d) sc += q[d] * Ks[k * SX + h * DH_ + d];
        sc *= scale;
      }
      mx = fmaxf(mx, sc);
    }
    float s = 0.f, acc[DH_];
#pragma unroll
    for (int d = 0; d < DH_; ++d) acc[d] = 0.f;
    for (int k = 0; k < N_; ++k) {
      float sc;
      if (mask[k] < 0.5f) sc = -1e9f;
      else {
        sc = 0.f;
#pragma unroll
        for (int d = 0; d < DH_; ++d) sc += q[d] * Ks[k * SX + h * DH_ + d];
        sc *= scale;
      }
      const float pz = __expf(sc - mx);
      s += pz;
#pragma unroll
      for (int d = 0; d < DH_; ++d) acc[d] += pz * Vs[k * SX + h * DH_ + d];
    }
    const float inv = 1.f / s;
#pragma unroll
    for (int d = 0; d < DH_; ++d) As[t * SX + h * DH_ + d] = acc[d] * inv;
  }
  __syncthreads();

  gemm_wmma(As, SX, p.Wo, p.bo, Qs, SX, 32, 32, false);
  __syncthreads();

  // residual + LayerNorm 1 (thread t owns row t)
  {
    float r[H_]; float m = 0.f;
#pragma unroll
    for (int hh = 0; hh < H_; ++hh) { r[hh] = Xs[t * SX + hh] + Qs[t * SX + hh]; m += r[hh]; }
    m *= (1.f / H_);
    float var = 0.f;
#pragma unroll
    for (int hh = 0; hh < H_; ++hh) { float d = r[hh] - m; var += d * d; }
    var *= (1.f / H_);
    const float inv = rsqrtf(var + 1e-5f);
#pragma unroll
    for (int hh = 0; hh < H_; ++hh)
      Xs[t * SX + hh] = (r[hh] - m) * inv * p.g1[hh] + p.lb1[hh];
  }
  __syncthreads();

  float* H1 = Qs;  // [256, SH] spans Qs..Ks (both free here)
  gemm_wmma(Xs, SX, p.Wf1, p.bf1, H1, SH, 32, 64, true);
  __syncthreads();
  gemm_wmma(H1, SH, p.Wf2, p.bf2, As, SX, 64, 32, false);
  __syncthreads();

  // residual + LayerNorm 2
  {
    float r[H_]; float m = 0.f;
#pragma unroll
    for (int hh = 0; hh < H_; ++hh) { r[hh] = Xs[t * SX + hh] + As[t * SX + hh]; m += r[hh]; }
    m *= (1.f / H_);
    float var = 0.f;
#pragma unroll
    for (int hh = 0; hh < H_; ++hh) { float d = r[hh] - m; var += d * d; }
    var *= (1.f / H_);
    const float inv = rsqrtf(var + 1e-5f);
#pragma unroll
    for (int hh = 0; hh < H_; ++hh)
      Xs[t * SX + hh] = (r[hh] - m) * inv * p.g2[hh] + p.lb2[hh];
  }
  __syncthreads();
}

// ---------------------------------------------------------------------------
// GAT pass 1: block=(b,i), thread=j. Recompute feat[b,i,j,:], per-head
// e = exp(leaky(aw))*valid, reduce over j -> S[(b*N+i)*NH + k] (+1e-9).
// ---------------------------------------------------------------------------
__global__ __launch_bounds__(256) void gat_denom_kernel(
    const float* __restrict__ feature, const float* __restrict__ valid,
    const float* __restrict__ W1, const float* __restrict__ b1,
    const float* __restrict__ Wml, const float* __restrict__ bml,
    float* __restrict__ S)
{
  __shared__ float w1s[IN_ * H_];
  __shared__ float red[NH_ * N_];
  const int j = threadIdx.x;
  const size_t bi = blockIdx.x;            // b*N + i
  if (j < IN_ * H_) w1s[j] = W1[j];
  __syncthreads();

  float xin[IN_];
  const float* fp = feature + (bi * N_ + j) * IN_;
#pragma unroll
  for (int c = 0; c < IN_; ++c) xin[c] = fp[c];
  float f[H_];
#pragma unroll
  for (int hh = 0; hh < H_; ++hh) {
    float acc = b1[hh];
#pragma unroll
    for (int c = 0; c < IN_; ++c) acc += xin[c] * w1s[c * H_ + hh];
    f[hh] = acc;
  }
  const float vld = valid[bi * N_ + j];
  for (int k = 0; k < NH_; ++k) {
    float aw = bml[k];
#pragma unroll
    for (int hh = 0; hh < H_; ++hh) aw += f[hh] * Wml[k * H_ + hh];
    const float l = aw >= 0.f ? aw : 0.01f * aw;
    red[k * N_ + j] = __expf(l) * vld;
  }
  __syncthreads();
  for (int s = 128; s >= 1; s >>= 1) {
    if (j < s)
      for (int k = 0; k < NH_; ++k) red[k * N_ + j] += red[k * N_ + j + s];
    __syncthreads();
  }
  if (j < NH_) S[bi * NH_ + j] = red[j * N_] + 1e-9f;
}

// ---------------------------------------------------------------------------
// GAT pass 2: block=(b,j), thread=i. hidden[b,j,h] = sum_i (mean_k w_k)*feat.
// ---------------------------------------------------------------------------
__global__ __launch_bounds__(256) void gat_agg_kernel(
    const float* __restrict__ feature, const float* __restrict__ valid,
    const float* __restrict__ W1, const float* __restrict__ b1,
    const float* __restrict__ Wml, const float* __restrict__ bml,
    const float* __restrict__ S, float* __restrict__ hidden)
{
  __shared__ float w1s[IN_ * H_];
  __shared__ float sacc[N_ * H_];
  const int i = threadIdx.x;
  const int b = blockIdx.x >> 8;
  const int j = blockIdx.x & 255;
  if (i < IN_ * H_) w1s[i] = W1[i];
  __syncthreads();

  const size_t base = ((size_t)(b * N_ + i)) * N_ + j;
  const float* fp = feature + base * IN_;
  float xin[IN_];
#pragma unroll
  for (int c = 0; c < IN_; ++c) xin[c] = fp[c];
  float f[H_];
#pragma unroll
  for (int hh = 0; hh < H_; ++hh) {
    float acc = b1[hh];
#pragma unroll
    for (int c = 0; c < IN_; ++c) acc += xin[c] * w1s[c * H_ + hh];
    f[hh] = acc;
  }
  const float vld = valid[base];
  float wsum = 0.f;
  for (int k = 0; k < NH_; ++k) {
    float aw = bml[k];
#pragma unroll
    for (int hh = 0; hh < H_; ++hh) aw += f[hh] * Wml[k * H_ + hh];
    const float l = aw >= 0.f ? aw : 0.01f * aw;
    const float e = __expf(l) * vld;
    wsum += e / S[(size_t)(b * N_ + i) * NH_ + k];
  }
  const float coeff = 0.25f * wsum;        // mean over heads folded in
#pragma unroll
  for (int hh = 0; hh < H_; ++hh) sacc[i * H_ + hh] = coeff * f[hh];
  __syncthreads();
  for (int s = 128; s >= 1; s >>= 1) {
    if (i < s)
      for (int hh = 0; hh < H_; ++hh) sacc[i * H_ + hh] += sacc[(i + s) * H_ + hh];
    __syncthreads();
  }
  if (i < H_) hidden[((size_t)blockIdx.x) * H_ + i] = sacc[i];
}

// ---------------------------------------------------------------------------
// Fused transformer: 1 block per batch, 8 waves, activations resident in LDS.
// enc0 -> enc1 -> (save trunk) -> policy layer -> probs; reload -> value layer.
// ---------------------------------------------------------------------------
__global__ __launch_bounds__(256) void net_kernel(
    const float* __restrict__ hidden, const float* __restrict__ valid,
    LayerParams e0, LayerParams e1, LayerParams pp, LayerParams vp,
    const float* __restrict__ Wpl, const float* __restrict__ bpl,
    const float* __restrict__ Wout, const float* __restrict__ bout,
    float* __restrict__ trunk, float* __restrict__ out)
{
  __shared__ float smem[5 * N_ * SX + 2 * N_];
  float* Xs   = smem;
  float* Qs   = Xs + N_ * SX;
  float* Ks   = Qs + N_ * SX;
  float* Vs   = Ks + N_ * SX;
  float* As   = Vs + N_ * SX;
  float* mask = As + N_ * SX;
  float* red  = mask + N_;

  const int b = blockIdx.x, t = threadIdx.x;

  for (int idx = t; idx < N_ * H_; idx += 256) {
    const int r = idx >> 5, c = idx & 31;
    Xs[r * SX + c] = hidden[(size_t)(b * N_ + r) * H_ + c];
  }
  mask[t] = valid[((size_t)(b * N_ + t)) * N_ + t];   // diagonal
  __syncthreads();

  encoder_layer(Xs, Qs, Ks, Vs, As, mask, e0);
  encoder_layer(Xs, Qs, Ks, Vs, As, mask, e1);

  // save trunk (each thread stores & later reloads its own addresses)
  float* tb = trunk + (size_t)b * N_ * H_;
  for (int idx = t; idx < N_ * H_; idx += 256) {
    const int r = idx >> 5, c = idx & 31;
    tb[idx] = Xs[r * SX + c];
  }

  // ---- policy branch (in place) ----
  encoder_layer(Xs, Qs, Ks, Vs, As, mask, pp);
  float lp = bpl[0];
#pragma unroll
  for (int hh = 0; hh < H_; ++hh) lp += Xs[t * SX + hh] * Wpl[hh];
  const float lg = 10.f * tanhf(lp) - 9999999999.0f * (1.f - mask[t]);
  red[t] = lg; __syncthreads();
  for (int s = 128; s >= 1; s >>= 1) {
    if (t < s) red[t] = fmaxf(red[t], red[t + s]);
    __syncthreads();
  }
  const float mx = red[0]; __syncthreads();
  const float ez = __expf(lg - mx);
  red[t] = ez; __syncthreads();
  for (int s = 128; s >= 1; s >>= 1) {
    if (t < s) red[t] += red[t + s];
    __syncthreads();
  }
  const float denom = red[0]; __syncthreads();
  out[b * N_ + t] = ez / denom;

  // ---- value branch ----
  for (int idx = t; idx < N_ * H_; idx += 256) {
    const int r = idx >> 5, c = idx & 31;
    Xs[r * SX + c] = tb[idx];
  }
  __syncthreads();
  encoder_layer(Xs, Qs, Ks, Vs, As, mask, vp);

#pragma unroll
  for (int hh = 0; hh < H_; ++hh) Qs[t * SX + hh] = Xs[t * SX + hh] * mask[t];
  __syncthreads();
  for (int s = 128; s >= 1; s >>= 1) {
    if (t < s)
      for (int hh = 0; hh < H_; ++hh) Qs[t * SX + hh] += Qs[(t + s) * SX + hh];
    __syncthreads();
  }
  if (t == 0) {
    float v = bout[0];
    for (int hh = 0; hh < H_; ++hh) v += Qs[hh] * Wout[hh];
    out[B_ * N_ + b] = v;
  }
}

// ---------------------------------------------------------------------------
static inline LayerParams mk(void* const* d, int o) {
  LayerParams p;
  p.Wq  = (const float*)d[o + 0];  p.bq  = (const float*)d[o + 1];
  p.Wk  = (const float*)d[o + 2];  p.bk  = (const float*)d[o + 3];
  p.Wv  = (const float*)d[o + 4];  p.bv  = (const float*)d[o + 5];
  p.Wo  = (const float*)d[o + 6];  p.bo  = (const float*)d[o + 7];
  p.Wf1 = (const float*)d[o + 8];  p.bf1 = (const float*)d[o + 9];
  p.Wf2 = (const float*)d[o + 10]; p.bf2 = (const float*)d[o + 11];
  p.g1  = (const float*)d[o + 12]; p.lb1 = (const float*)d[o + 13];
  p.g2  = (const float*)d[o + 14]; p.lb2 = (const float*)d[o + 15];
  return p;
}

extern "C" void kernel_launch(void* const* d_in, const int* in_sizes, int n_in,
                              void* d_out, int out_size, void* d_ws, size_t ws_size,
                              hipStream_t stream) {
  (void)in_sizes; (void)n_in; (void)out_size; (void)ws_size;
  const float* feature = (const float*)d_in[0];
  const float* valid   = (const float*)d_in[1];
  const float* W1      = (const float*)d_in[2];
  const float* b1      = (const float*)d_in[3];
  const float* Wml     = (const float*)d_in[4];
  const float* bml     = (const float*)d_in[5];
  const LayerParams e0 = mk(d_in, 6);
  const LayerParams e1 = mk(d_in, 22);
  const LayerParams pp = mk(d_in, 38);
  const LayerParams vp = mk(d_in, 54);
  const float* Wpl  = (const float*)d_in[70];
  const float* bpl  = (const float*)d_in[71];
  const float* Wout = (const float*)d_in[72];
  const float* bout = (const float*)d_in[73];

  float* S      = (float*)d_ws;                 // NH*B*N           = 16384 f
  float* hidden = S + (size_t)NH_ * B_ * N_;    // B*N*H            = 131072 f
  float* trunk  = hidden + (size_t)B_ * N_ * H_;// B*N*H            = 131072 f

  gat_denom_kernel<<<B_ * N_, 256, 0, stream>>>(feature, valid, W1, b1, Wml, bml, S);
  gat_agg_kernel<<<B_ * N_, 256, 0, stream>>>(feature, valid, W1, b1, Wml, bml, S, hidden);
  net_kernel<<<B_, 256, 0, stream>>>(hidden, valid, e0, e1, pp, vp,
                                     Wpl, bpl, Wout, bout, trunk, (float*)d_out);
}
